// OuterProduct_2482491097345
// MI455X (gfx1250) — matrix operand order repaired and checked by reading, not compiled, not run on hardware
//
#include <hip/hip_runtime.h>

// ---------------------------------------------------------------------------
// out = einsum("bti,btj,ijf->btf", x1, x2, K12) + x1@K1 + x2@K2 + bias
// M = B*T = 8192 rows, i=j=f=128.
//
// MI455X (gfx1250, wave32) mapping:
//   out[m,f] = sum_i (x1[m,i]*x2[m,:]) @ K12[i,:,:]   (+ linear terms)
// Per wave: M=32 (two 16-row subtiles; each B fragment feeds 2 WMMAs, halving
// B-operand bandwidth, the binding resource). x2 rows live in VGPRs as bf16 A
// fragments; per slice i they are rescaled by bf16(x1[m,i]) with
// v_pk_mul_bf16 (inline asm) and fed straight into v_wmma_f32_16x16x32_bf16,
// which accumulates into persistent f32 C/D registers -> no WMMA->VALU
// round-trip or hazard NOPs in the steady state.
// Slices 128/129 carry K2/K1 (scale=1.0) so linear terms ride the same loop.
// Split-K=4 across workgroups (1024 waves); deterministic reduce adds the
// three ws partials + bias.
// ---------------------------------------------------------------------------

typedef __attribute__((ext_vector_type(16))) __bf16 v16bf;
typedef __attribute__((ext_vector_type(8)))  __bf16 v8bf;
typedef __attribute__((ext_vector_type(8)))  float  v8f;
typedef __attribute__((ext_vector_type(8)))  unsigned int v8u;

#define NROWS   8192
#define NK      128
#define NF      128
#define NSLICE  130
#define FRAG_EL 512           // 32x16 bf16 fragment = 512 elements = 1KB
#define WAVE_M  32
#define WAVES_PER_BLOCK 4     // wg covers 128 rows
#define MBLOCKS 64
#define NSPLIT  4
#define BF16_ONE_X2 0x3F803F80u

// workspace byte offsets
#define WS_X1B  (0u)                     // x1 bf16, row-major (2 MB)
#define WS_X2B  (2u * 1024u * 1024u)     // x2 bf16, row-major (2 MB)
#define WS_X1P  (4u * 1024u * 1024u)     // x1 as duplicated bf16 pairs (4 MB)
#define WS_BF   (8u * 1024u * 1024u)     // 130*32 fragments * 1KB = 4.0625 MB
#define WS_P1   (13u * 1024u * 1024u)
#define WS_P2   (17u * 1024u * 1024u)
#define WS_P3   (21u * 1024u * 1024u)

// ---------------------------------------------------------------- pack x->bf16
__global__ void pack_x_bf16(const float* __restrict__ x,
                            unsigned short* __restrict__ xb_, int n) {
  __bf16* xb = (__bf16*)xb_;
  int idx = blockIdx.x * blockDim.x + threadIdx.x;
  if (idx < n) xb[idx] = (__bf16)x[idx];
}

// ---------------------------------------- pack x1 -> duplicated bf16 pair u32
__global__ void pack_x_dup(const float* __restrict__ x,
                           unsigned int* __restrict__ xp, int n) {
  int idx = blockIdx.x * blockDim.x + threadIdx.x;
  if (idx < n) {
    __bf16 b = (__bf16)x[idx];
    unsigned short u;
    __builtin_memcpy(&u, &b, 2);
    xp[idx] = (unsigned int)u | ((unsigned int)u << 16);
  }
}

// ------------------------------------------------ pack K12/K1/K2 -> B fragments
// Fragment (slice, kc, nt): 32x16 bf16 B operand for v_wmma_f32_16x16x32_bf16.
// Register layout (ISA 7.12.2): lane l: n = l%16 ; slot s: k = kc*32 + 8*(l>=16) + s + 8*(s>=8)
// Stored lane-major (l*32 bytes per lane) -> fully coalesced b128 pairs.
__global__ void pack_bfrag(const float* __restrict__ K12,
                           const float* __restrict__ K1,
                           const float* __restrict__ K2,
                           unsigned short* __restrict__ Bf_) {
  __bf16* Bf = (__bf16*)Bf_;
  int tid = blockIdx.x * blockDim.x + threadIdx.x;  // 130*4*8*32 = 133120
  int l     = tid & 31;
  int nt    = (tid >> 5) & 7;
  int kc    = (tid >> 8) & 3;
  int slice = tid >> 10;
  if (slice >= NSLICE) return;

  int ncol = nt * 16 + (l & 15);
  int j0   = kc * 32 + ((l & 16) ? 8 : 0);

  const float* src;
  if (slice < 128)       src = K12 + ((size_t)slice * NK + j0) * NF + ncol;
  else if (slice == 128) src = K2  + (size_t)j0 * NF + ncol;
  else                   src = K1  + (size_t)j0 * NF + ncol;

  v16bf v;
#pragma unroll
  for (int s = 0; s < 16; ++s) v[s] = (__bf16)src[(size_t)(s + ((s & 8) ? 8 : 0)) * NF];

  *(v16bf*)(Bf + (size_t)tid * 16) = v;
}

// A-operand fragments (16x32 bf16) from a plain row-major bf16 matrix.
// ISA A layout: lane l: m=l%16; slots 0..7 -> k=h8..h8+7, 8..15 -> 16+h8..23+h8.
__device__ __forceinline__ void load_a_frags(const __bf16* __restrict__ xrow,
                                             int h8, v16bf* __restrict__ A) {
#pragma unroll
  for (int kc = 0; kc < 4; ++kc) {
    v8bf lo = *(const v8bf*)(xrow + kc * 32 + h8);
    v8bf hi = *(const v8bf*)(xrow + kc * 32 + 16 + h8);
    A[kc] = __builtin_shufflevector(lo, hi, 0, 1, 2, 3, 4, 5, 6, 7,
                                    8, 9, 10, 11, 12, 13, 14, 15);
  }
}

// ---------------------------------------------------- bf16 packed scale (VALU)
__device__ __forceinline__ v16bf scale_bf16(v16bf a, unsigned int s) {
  v8u au, ru;
  __builtin_memcpy(&au, &a, 32);
#pragma unroll
  for (int d = 0; d < 8; ++d) {
    unsigned int t;
    asm("v_pk_mul_bf16 %0, %1, %2" : "=v"(t) : "v"(au[d]), "v"(s));
    ru[d] = t;
  }
  v16bf r;
  __builtin_memcpy(&r, &ru, 32);
  return r;
}

// --------------------------------------------- one slice, two 16-row subtiles
__device__ __forceinline__ void slice_mma2(const v16bf* __restrict__ A0,
                                           const v16bf* __restrict__ A1,
                                           unsigned int s0, unsigned int s1,
                                           const __bf16* __restrict__ fb,
                                           int lane,
                                           v8f* __restrict__ acc0,
                                           v8f* __restrict__ acc1) {
#pragma unroll
  for (int kc = 0; kc < 4; ++kc) {
    v16bf Ap0 = scale_bf16(A0[kc], s0);
    v16bf Ap1 = scale_bf16(A1[kc], s1);
#pragma unroll
    for (int nt = 0; nt < 8; ++nt) {
      v16bf bb = *(const v16bf*)(fb + (size_t)(kc * 8 + nt) * FRAG_EL + lane * 16);
      acc0[nt] = __builtin_amdgcn_wmma_f32_16x16x32_bf16(false, Ap0, false, bb,
                                                         (short)0, acc0[nt],
                                                         false, false);
      acc1[nt] = __builtin_amdgcn_wmma_f32_16x16x32_bf16(false, Ap1, false, bb,
                                                         (short)0, acc1[nt],
                                                         false, false);
    }
  }
}

// ------------------------------------------------------------------ main GEMM
__global__ __launch_bounds__(128) void bilinear_wmma(
    const unsigned short* __restrict__ x1b_,
    const unsigned short* __restrict__ x2b_,
    const unsigned int*  __restrict__ x1p,
    const unsigned short* __restrict__ Bf_,
    float* __restrict__ out0, float* __restrict__ p1,
    float* __restrict__ p2,  float* __restrict__ p3) {
  const __bf16* x1b = (const __bf16*)x1b_;
  const __bf16* x2b = (const __bf16*)x2b_;
  const __bf16* Bf  = (const __bf16*)Bf_;

  const int lane = threadIdx.x & 31;
  const int wave = threadIdx.x >> 5;
  const int ks   = blockIdx.x >> 6;          // K-split id 0..3
  const int mblk = blockIdx.x & (MBLOCKS - 1);
  const int mb   = (mblk * WAVES_PER_BLOCK + wave) * WAVE_M;

  const int h8 = (lane & 16) ? 8 : 0;        // half-select for A and C layouts

  // resident A (x2 rows, bf16) for both subtiles
  v16bf A2a[4], A2b[4];
  load_a_frags(x2b + (size_t)(mb      + (lane & 15)) * NK, h8, A2a);
  load_a_frags(x2b + (size_t)(mb + 16 + (lane & 15)) * NK, h8, A2b);

  v8f acc0[8], acc1[8];
#pragma unroll
  for (int nt = 0; nt < 8; ++nt)
#pragma unroll
    for (int v = 0; v < 8; ++v) { acc0[nt][v] = 0.f; acc1[nt][v] = 0.f; }

  // A-scale source: lane l scales rows m = base + (l & 15)
  const unsigned int* sp0 = x1p + (size_t)(mb      + (lane & 15)) * NK;
  const unsigned int* sp1 = x1p + (size_t)(mb + 16 + (lane & 15)) * NK;

  const int i0 = (ks == 0) ? 0 : (ks == 1) ? 33 : (ks == 2) ? 66 : 98;
  const int i1 = (ks == 0) ? 33 : (ks == 1) ? 66 : (ks == 2) ? 98 : 128;

  for (int i = i0; i < i1; ++i) {
    unsigned int s0 = sp0[i];
    unsigned int s1 = sp1[i];
    slice_mma2(A2a, A2b, s0, s1, Bf + (size_t)i * 32 * FRAG_EL, lane,
               acc0, acc1);
  }

  if (ks == 3) {  // linear terms: slice 128 (K2, A=x2), slice 129 (K1, A=x1)
    slice_mma2(A2a, A2b, BF16_ONE_X2, BF16_ONE_X2,
               Bf + (size_t)128 * 32 * FRAG_EL, lane, acc0, acc1);
    v16bf A1a[4], A1b[4];
    load_a_frags(x1b + (size_t)(mb      + (lane & 15)) * NK, h8, A1a);
    load_a_frags(x1b + (size_t)(mb + 16 + (lane & 15)) * NK, h8, A1b);
    slice_mma2(A1a, A1b, BF16_ONE_X2, BF16_ONE_X2,
               Bf + (size_t)129 * 32 * FRAG_EL, lane, acc0, acc1);
  }

  float* dst = (ks == 0) ? out0 : (ks == 1) ? p1 : (ks == 2) ? p2 : p3;
#pragma unroll
  for (int nt = 0; nt < 8; ++nt)
#pragma unroll
    for (int v = 0; v < 8; ++v) {
      dst[(size_t)(mb      + h8 + v) * NF + nt * 16 + (lane & 15)] = acc0[nt][v];
      dst[(size_t)(mb + 16 + h8 + v) * NF + nt * 16 + (lane & 15)] = acc1[nt][v];
    }
}

// ---------------------------------------------------- deterministic final add
__global__ void reduce_add(float* __restrict__ out,
                           const float* __restrict__ p1,
                           const float* __restrict__ p2,
                           const float* __restrict__ p3,
                           const float* __restrict__ bias, int n) {
  int idx = blockIdx.x * blockDim.x + threadIdx.x;
  if (idx < n)
    out[idx] = ((out[idx] + p1[idx]) + (p2[idx] + p3[idx])) + bias[idx & (NF - 1)];
}

// ---------------------------------------------------------------------- launch
extern "C" void kernel_launch(void* const* d_in, const int* in_sizes, int n_in,
                              void* d_out, int out_size, void* d_ws, size_t ws_size,
                              hipStream_t stream) {
  const float* x1   = (const float*)d_in[0];
  const float* x2   = (const float*)d_in[1];
  const float* K12  = (const float*)d_in[2];
  const float* K1   = (const float*)d_in[3];
  const float* K2   = (const float*)d_in[4];
  const float* bias = (const float*)d_in[5];
  float* out = (float*)d_out;

  char* ws = (char*)d_ws;
  unsigned short* x1b = (unsigned short*)(ws + WS_X1B);
  unsigned short* x2b = (unsigned short*)(ws + WS_X2B);
  unsigned int*   x1p = (unsigned int*)(ws + WS_X1P);
  unsigned short* Bf  = (unsigned short*)(ws + WS_BF);
  float* P1 = (float*)(ws + WS_P1);
  float* P2 = (float*)(ws + WS_P2);
  float* P3 = (float*)(ws + WS_P3);

  const int nx = NROWS * NK;  // 1,048,576
  pack_x_bf16<<<nx / 256, 256, 0, stream>>>(x1, x1b, nx);
  pack_x_bf16<<<nx / 256, 256, 0, stream>>>(x2, x2b, nx);
  pack_x_dup<<<nx / 256, 256, 0, stream>>>(x1, x1p, nx);
  pack_bfrag<<<(NSLICE * 1024) / 256, 256, 0, stream>>>(K12, K1, K2, Bf);

  bilinear_wmma<<<MBLOCKS * NSPLIT, WAVES_PER_BLOCK * 32, 0, stream>>>(
      x1b, x2b, x1p, Bf, out, P1, P2, P3);

  reduce_add<<<(NROWS * NF) / 256, 256, 0, stream>>>(out, P1, P2, P3, bias,
                                                     NROWS * NF);
}
